// BasicLSTMClassifierWithAttention_56547539419802
// MI455X (gfx1250) — matrix operand order repaired and checked by reading, not compiled, not run on hardware
//
#include <hip/hip_runtime.h>
#include <hip/hip_bf16.h>

// ---------------------------------------------------------------------------
// MI455X (gfx1250) implementation: f16 WMMA GEMMs with f32 accumulation.
//   v_wmma_f32_16x16x32_f16 throughout; persistent grid-synced LSTM scan with
//   w_hh held entirely in VGPRs and gate-interleaved xw for b64 epilogue loads.
// ---------------------------------------------------------------------------

typedef __attribute__((ext_vector_type(16))) _Float16 v16h;
typedef __attribute__((ext_vector_type(8)))  _Float16 v8h;
typedef __attribute__((ext_vector_type(4)))  _Float16 v4h;
typedef __attribute__((ext_vector_type(8)))  float    v8f;

#define BB   512
#define CC   271
#define TT   281
#define HH   128
#define NCLS 1854
#define CP   288      // C padded to multiple of 32
#define G4   512      // 4*H
#define BT   (BB*TT)  // 143872
#define NP   1856     // NCLS padded to multiple of 16

// ---- WMMA tile loaders (per CDNA5 ISA 7.12.2 VGPR layouts) ----------------
// A 16x32 f16 tile from row-major [M,K]:
//   lane l<16 : row l, K {0..7} then {16..23}; lane l>=16: row l-16, K {8..15},{24..31}
__device__ __forceinline__ v16h load_tileA(const _Float16* base, int ld, int lane) {
  const int r = lane & 15, hf = lane >> 4;
  const _Float16* p = base + (size_t)r * ld + hf * 8;
  v8h lo = *(const v8h*)p;
  v8h hi = *(const v8h*)(p + 16);
  v16h o;
#pragma unroll
  for (int i = 0; i < 8; ++i) { o[i] = lo[i]; o[8 + i] = hi[i]; }
  return o;
}
// B 32x16 f16 tile from row-major [N,K] weights (lane n = output column n):
//   lane n<16 : row n, K {0..7},{16..23} ; lane n>=16 : row n-16, K {8..15},{24..31}
__device__ __forceinline__ v16h load_tileB(const _Float16* base, int ld, int lane) {
  const int n = lane & 15, hf = lane >> 4;
  const _Float16* p = base + (size_t)n * ld + hf * 8;
  v8h lo = *(const v8h*)p;
  v8h hi = *(const v8h*)(p + 16);
  v16h o;
#pragma unroll
  for (int i = 0; i < 8; ++i) { o[i] = lo[i]; o[8 + i] = hi[i]; }
  return o;
}
#define WMMA_F16(a, b, c) \
  __builtin_amdgcn_wmma_f32_16x16x32_f16(false, (a), false, (b), (short)0, (c), false, false)

__device__ __forceinline__ float sigf(float x) { return 1.0f / (1.0f + __expf(-x)); }

// ---- utility kernels -------------------------------------------------------
__global__ void k_zero(unsigned int* p, long n) {
  long i = (long)blockIdx.x * blockDim.x + threadIdx.x;
  if (i < n) p[i] = 0u;
}

// fp32 [N,K] -> f16 [Npad,Kpad] (zero padded)
__global__ void k_pack(const float* __restrict__ src, _Float16* __restrict__ dst,
                       int N, int K, int Npad, int Kpad) {
  long i = (long)blockIdx.x * blockDim.x + threadIdx.x;
  long tot = (long)Npad * Kpad;
  if (i >= tot) return;
  int n = (int)(i / Kpad), k = (int)(i % Kpad);
  float v = (n < N && k < K) ? src[(size_t)n * K + k] : 0.0f;
  dst[i] = (_Float16)v;
}

// att_W [2H,2H] fp32 -> transposed f16 [N=2H, K=2H]
__global__ void k_packT(const float* __restrict__ src, _Float16* __restrict__ dst) {
  int i = blockIdx.x * blockDim.x + threadIdx.x; // 256*256
  int n = i >> 8, k = i & 255;
  dst[i] = (_Float16)src[k * 256 + n];
}

// bsum[dirlayer][512] = b_ih + b_hh   (0=l0f 1=l0b 2=l1f 3=l1b)
__global__ void k_bsum(const float* a0, const float* b0, const float* a1, const float* b1,
                       const float* a2, const float* b2, const float* a3, const float* b3,
                       float* __restrict__ out) {
  int i = blockIdx.x * blockDim.x + threadIdx.x; // 4*512
  int w = i >> 9, j = i & 511;
  const float* A = (w == 0) ? a0 : (w == 1) ? a1 : (w == 2) ? a2 : a3;
  const float* Bp = (w == 0) ? b0 : (w == 1) ? b1 : (w == 2) ? b2 : b3;
  out[i] = A[j] + Bp[j];
}

// X [B,C,T] fp32 -> A0 [B,T,CP] f16 (LDS-tiled transpose, zero pad C)
__global__ void k_xT(const float* __restrict__ X, _Float16* __restrict__ A0) {
  __shared__ float tile[32][33];
  int b = blockIdx.z;
  int t0 = blockIdx.x * 32, c0 = blockIdx.y * 32;
  int tx = threadIdx.x, ty = threadIdx.y; // (32,8)
#pragma unroll
  for (int i = 0; i < 4; ++i) {
    int c = c0 + ty + i * 8, t = t0 + tx;
    float v = (c < CC && t < TT) ? X[((size_t)b * CC + c) * TT + t] : 0.0f;
    tile[ty + i * 8][tx] = v; // [c_local][t_local]
  }
  __syncthreads();
#pragma unroll
  for (int i = 0; i < 4; ++i) {
    int t = t0 + ty + i * 8, c = c0 + tx;
    if (t < TT && c < CP)
      A0[((size_t)b * TT + t) * CP + c] = (_Float16)tile[tx][ty + i * 8];
  }
}

// ---- input-projection GEMM: xw = A0 @ W^T + bsum  (both dirs) --------------
// Output gate-interleaved: xw[t][b][h][gate] so the recurrent epilogue reads
// all 4 gate pre-activations for one (b,h) with a single aligned b64 load.
// one wave per 16x16 tile; tiles = 8992 Mtiles * 64 (32 Ntiles * 2 dirs)
__global__ void k_gemm_xw(const _Float16* __restrict__ A0, int lda, int K,
                          const _Float16* __restrict__ Wf, const _Float16* __restrict__ Wb,
                          const float* __restrict__ bsf, const float* __restrict__ bsb,
                          _Float16* __restrict__ xwf, _Float16* __restrict__ xwb) {
  const int lane = threadIdx.x & 31;
  const long wid = (long)blockIdx.x * 8 + (threadIdx.x >> 5);
  const int nt64 = (int)(wid & 63);
  const long mt = wid >> 6; // 0..8991
  const int dir = nt64 >> 5;
  const int n0 = (nt64 & 31) * 16;
  const _Float16* W = dir ? Wb : Wf;
  const float* bs = dir ? bsb : bsf;
  _Float16* xw = dir ? xwb : xwf;
  const _Float16* Ab = A0 + (size_t)(mt * 16) * lda;
  const _Float16* Bb = W + (size_t)n0 * K;
  v8f acc = {};
  for (int k = 0; k < K; k += 32) {
    v16h a = load_tileA(Ab + k, lda, lane);
    v16h b = load_tileB(Bb + k, K, lane);
    acc = WMMA_F16(a, b, acc);
  }
  const int col = n0 + (lane & 15);           // 0..511 within this direction
  const int gilv = (col & 127) * 4 + (col >> 7); // gate-interleaved position
  const float bias = bs[col];
  const int rbase = (int)(mt * 16) + 8 * (lane >> 4);
#pragma unroll
  for (int v = 0; v < 8; ++v) {
    int r = rbase + v;
    int b_ = r / TT, t_ = r % TT;
    xw[((size_t)t_ * BB + b_) * G4 + gilv] = (_Float16)(acc[v] + bias);
  }
}

// ---- persistent recurrent scan (both directions of one layer) --------------
// 64 blocks x 256 threads; blocks 0..31 forward chain, 32..63 backward chain.
// Each wave owns a (16 batch rows x 16 h cols) tile and all 4 gate tiles;
// cell state AND all 16 w_hh B-tiles (128 VGPRs) live in registers for the
// whole 281-step scan -> per step only 8 A-tile loads + 8 b64 xw loads.
__global__ void k_lstm(const _Float16* __restrict__ whhF, const _Float16* __restrict__ whhB,
                       const _Float16* __restrict__ xwF, const _Float16* __restrict__ xwB,
                       _Float16* __restrict__ hbuf, // [chain][parity][B*H] f16
                       _Float16* __restrict__ hs,   // [B][T][2H] f16
                       int* __restrict__ counter) {
  const int lane = threadIdx.x & 31;
  const int wave = threadIdx.x >> 5;
  const int chain = blockIdx.x >> 5;     // 0 fwd, 1 bwd
  const int blk = blockIdx.x & 31;
  const int wc = blk * 8 + wave;         // 0..255
  const int m0 = (wc >> 3) * 16;         // batch rows
  const int c0 = (wc & 7) * 16;          // h cols
  const _Float16* whh = chain ? whhB : whhF;
  const _Float16* xw = chain ? xwB : xwF;
  _Float16* hb = hbuf + (size_t)chain * (2 * BB * HH);
  const int hcol = c0 + (lane & 15);
  const int rb = m0 + 8 * (lane >> 4);
  // preload w_hh: 4 gates x 4 K-steps of 32x16 B tiles, resident in VGPRs
  v16h bw[16];
#pragma unroll
  for (int g = 0; g < 4; ++g)
#pragma unroll
    for (int k8 = 0; k8 < 4; ++k8)
      bw[g * 4 + k8] = load_tileB(whh + (size_t)(g * HH + c0) * HH + k8 * 32, HH, lane);
  v8f cst = {};
  for (int s = 0; s < TT; ++s) {
    const int t = chain ? (TT - 1 - s) : s;
    const _Float16* hr = hb + (size_t)(s & 1) * (BB * HH);
    _Float16* hw = hb + (size_t)((s + 1) & 1) * (BB * HH);
    v8f a0 = {}, a1 = {}, a2 = {}, a3 = {};
#pragma unroll
    for (int k8 = 0; k8 < 4; ++k8) {
      v16h a = load_tileA(hr + (size_t)m0 * HH + k8 * 32, HH, lane);
      a0 = WMMA_F16(a, bw[0 * 4 + k8], a0);
      a1 = WMMA_F16(a, bw[1 * 4 + k8], a1);
      a2 = WMMA_F16(a, bw[2 * 4 + k8], a2);
      a3 = WMMA_F16(a, bw[3 * 4 + k8], a3);
    }
    const _Float16* xwt = xw + (size_t)t * BB * G4;
#pragma unroll
    for (int v = 0; v < 8; ++v) {
      int brow = rb + v;
      v4h gv = *(const v4h*)(xwt + (size_t)brow * G4 + hcol * 4); // i,f,g,o
      float gi = a0[v] + (float)gv[0];
      float gf = a1[v] + (float)gv[1];
      float gg = a2[v] + (float)gv[2];
      float go = a3[v] + (float)gv[3];
      float c = sigf(gf) * cst[v] + sigf(gi) * tanhf(gg);
      cst[v] = c;
      float hval = sigf(go) * tanhf(c);
      _Float16 hhh = (_Float16)hval;
      hw[(size_t)brow * HH + hcol] = hhh;
      hs[((size_t)brow * TT + t) * 256 + chain * HH + hcol] = hhh;
    }
    // one grid-wide barrier per timestep (release/acquire atomic counter)
    __syncthreads();
    if (threadIdx.x == 0) {
      __hip_atomic_fetch_add(counter, 1, __ATOMIC_RELEASE, __HIP_MEMORY_SCOPE_AGENT);
      const int target = 64 * (s + 1);
      while (__hip_atomic_load(counter, __ATOMIC_ACQUIRE, __HIP_MEMORY_SCOPE_AGENT) < target)
        __builtin_amdgcn_s_sleep(2);
    }
    __syncthreads();
  }
}

// ---- attention scores: a[b,t] = sum_j tanh(h1 @ attW)[j] * v[j] ------------
// one wave per 16-row tile; 8992 tiles.
__global__ void k_att(const _Float16* __restrict__ hs1, const _Float16* __restrict__ attWt,
                      const float* __restrict__ attv, float* __restrict__ aout) {
  const int lane = threadIdx.x & 31;
  const long wid = (long)blockIdx.x * 8 + (threadIdx.x >> 5); // 0..8991
  const _Float16* Ab = hs1 + (size_t)(wid * 16) * 256;
  v16h at[8];
#pragma unroll
  for (int k8 = 0; k8 < 8; ++k8) at[k8] = load_tileA(Ab + k8 * 32, 256, lane);
  float psum[8];
#pragma unroll
  for (int v = 0; v < 8; ++v) psum[v] = 0.0f;
  for (int nt = 0; nt < 16; ++nt) {
    v8f acc = {};
    const _Float16* Bb = attWt + (size_t)(nt * 16) * 256;
#pragma unroll
    for (int k8 = 0; k8 < 8; ++k8) acc = WMMA_F16(at[k8], load_tileB(Bb + k8 * 32, 256, lane), acc);
    float vv = attv[nt * 16 + (lane & 15)];
#pragma unroll
    for (int v = 0; v < 8; ++v) psum[v] += vv * tanhf(acc[v]);
  }
#pragma unroll
  for (int v = 0; v < 8; ++v) {
    float sv = psum[v];
    for (int off = 1; off < 16; off <<= 1) sv += __shfl_xor(sv, off, 16);
    psum[v] = sv;
  }
  if ((lane & 15) == 0) {
    int rbase = (int)(wid * 16) + 8 * (lane >> 4);
#pragma unroll
    for (int v = 0; v < 8; ++v) aout[rbase + v] = psum[v]; // row = b*T + t
  }
}

__global__ void k_softmax(const float* __restrict__ a, float* __restrict__ s) {
  __shared__ float red[256];
  int b = blockIdx.x, tid = threadIdx.x;
  const float* ap = a + (size_t)b * TT;
  float m = -3.4e38f;
  for (int t = tid; t < TT; t += 256) m = fmaxf(m, ap[t]);
  red[tid] = m; __syncthreads();
  for (int o = 128; o > 0; o >>= 1) { if (tid < o) red[tid] = fmaxf(red[tid], red[tid + o]); __syncthreads(); }
  float mx = red[0]; __syncthreads();
  float sum = 0.0f;
  for (int t = tid; t < TT; t += 256) sum += __expf(ap[t] - mx);
  red[tid] = sum; __syncthreads();
  for (int o = 128; o > 0; o >>= 1) { if (tid < o) red[tid] += red[tid + o]; __syncthreads(); }
  float inv = 1.0f / red[0];
  for (int t = tid; t < TT; t += 256) s[(size_t)b * TT + t] = __expf(ap[t] - mx) * inv;
}

// weighted[b,h] = sum_t scores[b,t] * h1[b,t,h]  -> f16
__global__ void k_weighted(const float* __restrict__ sc, const _Float16* __restrict__ hs1,
                           _Float16* __restrict__ wgt) {
  int b = blockIdx.x, h = threadIdx.x; // 256 threads
  const float* s = sc + (size_t)b * TT;
  const _Float16* hp = hs1 + (size_t)b * TT * 256 + h;
  float acc = 0.0f;
  for (int t = 0; t < TT; ++t) acc += s[t] * (float)hp[(size_t)t * 256];
  wgt[b * 256 + h] = (_Float16)acc;
}

// logits = weighted @ headW^T + head_b ; 32 Mtiles x 116 Ntiles = 3712 tiles
__global__ void k_head(const _Float16* __restrict__ wgt, const _Float16* __restrict__ hW,
                       const float* __restrict__ hb, float* __restrict__ out) {
  const int lane = threadIdx.x & 31;
  const int wid = blockIdx.x * 8 + (threadIdx.x >> 5); // 0..3711
  const int mt = wid / 116, nt = wid % 116;
  const int m0 = mt * 16, n0 = nt * 16;
  v8f acc = {};
#pragma unroll
  for (int k = 0; k < 256; k += 32) {
    v16h a = load_tileA(wgt + (size_t)m0 * 256 + k, 256, lane);
    v16h b = load_tileB(hW + (size_t)n0 * 256 + k, 256, lane);
    acc = WMMA_F16(a, b, acc);
  }
  const int col = n0 + (lane & 15);
  if (col < NCLS) {
    float bias = hb[col];
    int rbase = m0 + 8 * (lane >> 4);
#pragma unroll
    for (int v = 0; v < 8; ++v) out[(size_t)(rbase + v) * NCLS + col] = acc[v] + bias;
  }
}

// ---------------------------------------------------------------------------
extern "C" void kernel_launch(void* const* d_in, const int* in_sizes, int n_in,
                              void* d_out, int out_size, void* d_ws, size_t ws_size,
                              hipStream_t stream) {
  const float* X      = (const float*)d_in[0];
  const float* wih0f  = (const float*)d_in[1];
  const float* whh0f  = (const float*)d_in[2];
  const float* bih0f  = (const float*)d_in[3];
  const float* bhh0f  = (const float*)d_in[4];
  const float* wih0b  = (const float*)d_in[5];
  const float* whh0b  = (const float*)d_in[6];
  const float* bih0b  = (const float*)d_in[7];
  const float* bhh0b  = (const float*)d_in[8];
  const float* wih1f  = (const float*)d_in[9];
  const float* whh1f  = (const float*)d_in[10];
  const float* bih1f  = (const float*)d_in[11];
  const float* bhh1f  = (const float*)d_in[12];
  const float* wih1b  = (const float*)d_in[13];
  const float* whh1b  = (const float*)d_in[14];
  const float* bih1b  = (const float*)d_in[15];
  const float* bhh1b  = (const float*)d_in[16];
  const float* attW   = (const float*)d_in[17];
  const float* attv   = (const float*)d_in[18];
  const float* headW  = (const float*)d_in[19];
  const float* headb  = (const float*)d_in[20];
  float* out = (float*)d_out;

  // ---- workspace layout ----
  char* ws = (char*)d_ws;
  size_t off = 0;
  auto alloc = [&](size_t bytes) { size_t o = off; off = (off + bytes + 255) & ~(size_t)255; return o; };
  size_t oA0   = alloc((size_t)BT * CP * 2);        // X transposed+padded f16
  size_t oXWf  = alloc((size_t)TT * BB * G4 * 2);   // xw forward  [T,B,H,4] f16
  size_t oXWb  = alloc((size_t)TT * BB * G4 * 2);   // xw backward
  size_t oHS0  = alloc((size_t)BT * 256 * 2);       // layer0 outputs [B,T,2H] f16
  size_t oHS1  = alloc((size_t)BT * 256 * 2);       // layer1 outputs
  size_t oHB   = alloc((size_t)2 * 2 * 2 * BB * HH * 2); // hbuf [layer][chain][parity]
  size_t oCNT  = alloc(256);                        // 2 barrier counters
  size_t oA    = alloc((size_t)BT * 4);             // attention logits f32
  size_t oSC   = alloc((size_t)BT * 4);             // softmax scores
  size_t oWGT  = alloc((size_t)BB * 256 * 2);       // weighted f16
  size_t oPI0f = alloc((size_t)G4 * CP * 2);        // packed weights (f16)
  size_t oPI0b = alloc((size_t)G4 * CP * 2);
  size_t oPI1f = alloc((size_t)G4 * 256 * 2);
  size_t oPI1b = alloc((size_t)G4 * 256 * 2);
  size_t oPH0f = alloc((size_t)G4 * HH * 2);
  size_t oPH0b = alloc((size_t)G4 * HH * 2);
  size_t oPH1f = alloc((size_t)G4 * HH * 2);
  size_t oPH1b = alloc((size_t)G4 * HH * 2);
  size_t oPAT  = alloc((size_t)256 * 256 * 2);
  size_t oPHW  = alloc((size_t)NP * 256 * 2);
  size_t oBS   = alloc((size_t)4 * 512 * 4);
  (void)ws_size; (void)in_sizes; (void)n_in; (void)out_size;

  _Float16* A0   = (_Float16*)(ws + oA0);
  _Float16* XWf  = (_Float16*)(ws + oXWf);
  _Float16* XWb  = (_Float16*)(ws + oXWb);
  _Float16* HS0  = (_Float16*)(ws + oHS0);
  _Float16* HS1  = (_Float16*)(ws + oHS1);
  _Float16* HB   = (_Float16*)(ws + oHB);
  int*      CNT  = (int*)(ws + oCNT);
  float*    Aat  = (float*)(ws + oA);
  float*    SC   = (float*)(ws + oSC);
  _Float16* WGT  = (_Float16*)(ws + oWGT);
  _Float16* PI0f = (_Float16*)(ws + oPI0f);
  _Float16* PI0b = (_Float16*)(ws + oPI0b);
  _Float16* PI1f = (_Float16*)(ws + oPI1f);
  _Float16* PI1b = (_Float16*)(ws + oPI1b);
  _Float16* PH0f = (_Float16*)(ws + oPH0f);
  _Float16* PH0b = (_Float16*)(ws + oPH0b);
  _Float16* PH1f = (_Float16*)(ws + oPH1f);
  _Float16* PH1b = (_Float16*)(ws + oPH1b);
  _Float16* PAT  = (_Float16*)(ws + oPAT);
  _Float16* PHW  = (_Float16*)(ws + oPHW);
  float*    BS   = (float*)(ws + oBS);

  // 1) zero hbuf + counters (must happen every call: graph replay safe)
  {
    long nh = (long)(2 * 2 * 2 * BB * HH * 2) / 4;
    k_zero<<<(unsigned)((nh + 255) / 256), 256, 0, stream>>>((unsigned int*)HB, nh);
    k_zero<<<1, 64, 0, stream>>>((unsigned int*)CNT, 64);
  }
  // 2) pack weights to f16
  auto packLaunch = [&](const float* s, _Float16* d, int N, int K, int Npad, int Kpad) {
    long tot = (long)Npad * Kpad;
    k_pack<<<(unsigned)((tot + 255) / 256), 256, 0, stream>>>(s, d, N, K, Npad, Kpad);
  };
  packLaunch(wih0f, PI0f, G4, CC, G4, CP);
  packLaunch(wih0b, PI0b, G4, CC, G4, CP);
  packLaunch(wih1f, PI1f, G4, 256, G4, 256);
  packLaunch(wih1b, PI1b, G4, 256, G4, 256);
  packLaunch(whh0f, PH0f, G4, HH, G4, HH);
  packLaunch(whh0b, PH0b, G4, HH, G4, HH);
  packLaunch(whh1f, PH1f, G4, HH, G4, HH);
  packLaunch(whh1b, PH1b, G4, HH, G4, HH);
  packLaunch(headW, PHW, NCLS, 256, NP, 256);
  k_packT<<<256, 256, 0, stream>>>(attW, PAT);
  k_bsum<<<8, 256, 0, stream>>>(bih0f, bhh0f, bih0b, bhh0b, bih1f, bhh1f, bih1b, bhh1b, BS);
  // 3) transpose X
  {
    dim3 g(9, 9, BB), blk(32, 8);
    k_xT<<<g, blk, 0, stream>>>(X, A0);
  }
  // 4) layer0 input projection (K=288 padded)  : 8992*64 tiles / 8 waves
  k_gemm_xw<<<71936, 256, 0, stream>>>(A0, CP, CP, PI0f, PI0b, BS + 0, BS + 512, XWf, XWb);
  // 5) layer0 recurrence (both directions)
  k_lstm<<<64, 256, 0, stream>>>(PH0f, PH0b, XWf, XWb, HB, HS0, CNT);
  // 6) layer1 input projection (A = HS0, K=256)
  k_gemm_xw<<<71936, 256, 0, stream>>>(HS0, 256, 256, PI1f, PI1b, BS + 1024, BS + 1536, XWf, XWb);
  // 7) layer1 recurrence
  k_lstm<<<64, 256, 0, stream>>>(PH1f, PH1b, XWf, XWb, HB + (size_t)2 * 2 * BB * HH, HS1, CNT + 64);
  // 8) attention scores (8992 tiles / 8 waves)
  k_att<<<1124, 256, 0, stream>>>(HS1, PAT, attv, Aat);
  // 9) softmax over T per batch row
  k_softmax<<<BB, 256, 0, stream>>>(Aat, SC);
  // 10) weighted sum over time
  k_weighted<<<BB, 256, 0, stream>>>(SC, HS1, WGT);
  // 11) classification head (3712 tiles / 8 waves)
  k_head<<<464, 256, 0, stream>>>(WGT, PHW, headb, out);
}
